// MaskedMultiHeadSelfAttention_7052336300492
// MI455X (gfx1250) — matrix operand order repaired and checked by reading, not compiled, not run on hardware
//
#include <hip/hip_runtime.h>
#include <stdint.h>

// ---------------------------------------------------------------------------
// Masked multi-head self-attention, CDNA5 (gfx1250), wave32 WMMA bf16 path
// with async global->LDS staging (ASYNCcnt) and double buffering.
// B=4, T=2048, D=1024, H=16, head_dim=64.
// ---------------------------------------------------------------------------

#define BB 4
#define TT 2048
#define DD 1024
#define HH 16
#define KH 64          // head dim

typedef __attribute__((ext_vector_type(16))) __bf16 v16bf;
typedef __attribute__((ext_vector_type(8)))  __bf16 v8bf;
typedef __attribute__((ext_vector_type(8)))  float  v8f;

// ---- helpers --------------------------------------------------------------

static __device__ __forceinline__ __bf16 f2bf(float f) {
    union { float f; uint32_t u; } x; x.f = f;
    uint32_t r = (x.u + 0x7FFFu + ((x.u >> 16) & 1u)) >> 16;   // RNE
    union { uint16_t s; __bf16 b; } y; y.s = (uint16_t)r;
    return y.b;
}

static __device__ __forceinline__ v8bf load8(const __bf16* p) {
    return *reinterpret_cast<const v8bf*>(p);          // 16B vector load
}
static __device__ __forceinline__ v16bf load16(const __bf16* p) {
    return *reinterpret_cast<const v16bf*>(p);         // 32B vector load
}
static __device__ __forceinline__ v16bf combine(v8bf lo, v8bf hi) {
    v16bf r;
#pragma unroll
    for (int i = 0; i < 8; ++i) { r[i] = lo[i]; r[i + 8] = hi[i]; }
    return r;
}
static __device__ __forceinline__ v8f zero8() {
    v8f c;
#pragma unroll
    for (int i = 0; i < 8; ++i) c[i] = 0.f;
    return c;
}
static __device__ __forceinline__ v8f wmma_bf16(v16bf a, v16bf b, v8f c) {
    return __builtin_amdgcn_wmma_f32_16x16x32_bf16(
        false, a, false, b, (short)0, c, false, false);
}

// Generic (flat) pointer to a __shared__ object: low 32 bits are the LDS
// byte offset.
static __device__ __forceinline__ uint32_t lds_off(const void* p) {
    return (uint32_t)(uintptr_t)p;
}

// Async copy: 16B per lane, global -> LDS, tracked by ASYNCcnt.
static __device__ __forceinline__ void async_cp16(uint32_t ldsa, const void* g) {
    asm volatile("global_load_async_to_lds_b128 %0, %1, off"
                 :: "v"(ldsa), "v"(g) : "memory");
}

#if __has_builtin(__builtin_amdgcn_s_wait_asynccnt)
#define WAIT_ASYNC(n) __builtin_amdgcn_s_wait_asynccnt(n)
#else
#define WAIT_ASYNC(n) asm volatile("s_wait_asynccnt %0" :: "i"(n) : "memory")
#endif

// ---- kernel 0: fp32 -> bf16 cast -----------------------------------------

__global__ void cast_f32_bf16(const float* __restrict__ src,
                              __bf16* __restrict__ dst, int n) {
    int i = blockIdx.x * blockDim.x + threadIdx.x;
    int stride = gridDim.x * blockDim.x;
    for (; i < n; i += stride) dst[i] = f2bf(src[i]);
}

// ---- kernel 1: fused QKV projection  Y = x @ W^T --------------------------
// Block tile: M=128 (8 waves x 16), N=64, K staged 64 at a time, double
// buffered in LDS via async copies. Q,K stored [B,H,T,64]; V transposed
// [B,H,64,T] so the attention PV B-fragment is contiguous.

__global__ __launch_bounds__(256)
void qkv_proj(const __bf16* __restrict__ xb,
              const __bf16* __restrict__ Wqb,
              const __bf16* __restrict__ Wkb,
              const __bf16* __restrict__ Wvb,
              __bf16* __restrict__ Qw,
              __bf16* __restrict__ Kw,
              __bf16* __restrict__ Vt) {
    __shared__ alignas(128) __bf16 As[2][128][64];   // 32 KB
    __shared__ alignas(128) __bf16 Bs[2][64][64];    // 16 KB

    const int tid  = threadIdx.x;
    const int lane = tid & 31;
    const int wave = tid >> 5;
    const int hi = lane >> 4;
    const int ln = lane & 15;
    const int z = blockIdx.z;
    const __bf16* __restrict__ Wb = (z == 0) ? Wqb : (z == 1) ? Wkb : Wvb;

    const int mBlock = blockIdx.y * 128;
    const int nBase  = blockIdx.x * 64;

    // Stage one K-slab (64 wide): A tile 128x64 (1024 16B chunks) +
    // B tile 64x64 (512 chunks) = 6 chunks per thread.
    auto issue = [&](int k0, int buf) {
#pragma unroll
        for (int j = 0; j < 4; ++j) {                 // A chunks
            const int c = tid + j * 256;
            const int row = c >> 3, off = (c & 7) * 8;
            async_cp16(lds_off(&As[buf][row][off]),
                       xb + (size_t)(mBlock + row) * DD + k0 + off);
        }
#pragma unroll
        for (int j = 0; j < 2; ++j) {                 // B chunks
            const int c = tid + j * 256;
            const int row = c >> 3, off = (c & 7) * 8;
            async_cp16(lds_off(&Bs[buf][row][off]),
                       Wb + (size_t)(nBase + row) * DD + k0 + off);
        }
    };

    v8f acc[4];
#pragma unroll
    for (int t = 0; t < 4; ++t) acc[t] = zero8();

    issue(0, 0);
    for (int s = 0; s < DD / 64; ++s) {
        const int cur = s & 1;
        if (s + 1 < DD / 64) { issue((s + 1) * 64, cur ^ 1); WAIT_ASYNC(6); }
        else                 { WAIT_ASYNC(0); }
        __syncthreads();
#pragma unroll
        for (int ks = 0; ks < 64; ks += 32) {
            const __bf16* arow = &As[cur][wave * 16 + ln][0];
            v16bf a = load16(arow + ks + hi * 16);
#pragma unroll
            for (int t = 0; t < 4; ++t) {
                v16bf bf = load16(&Bs[cur][t * 16 + ln][ks + hi * 16]);
                acc[t] = wmma_bf16(a, bf, acc[t]);
            }
        }
        __syncthreads();
    }

    const int mBase = mBlock + wave * 16;
#pragma unroll
    for (int t = 0; t < 4; ++t) {
#pragma unroll
        for (int r = 0; r < 8; ++r) {
            const int mG = mBase + r + 8 * hi;
            const int nG = nBase + t * 16 + ln;
            const int b = mG >> 11, tt = mG & (TT - 1);
            const int h = nG >> 6, c = nG & 63;
            const __bf16 v = f2bf(acc[t][r]);
            if (z == 0)
                Qw[((((size_t)b * HH + h) * TT + tt) << 6) + c] = v;
            else if (z == 1)
                Kw[((((size_t)b * HH + h) * TT + tt) << 6) + c] = v;
            else
                Vt[(((size_t)b * HH + h) * KH + c) * TT + tt] = v;
        }
    }
}

// ---- kernel 2: causal flash attention -------------------------------------
// One block = 8 waves = 128 consecutive q rows of one (b,h). K/V panels are
// staged in LDS (double buffered, async) and shared by all 8 waves.

__global__ __launch_bounds__(256)
void flash_attn(const __bf16* __restrict__ Qw,
                const __bf16* __restrict__ Kw,
                const __bf16* __restrict__ Vt,
                __bf16* __restrict__ Ow) {
    __shared__ alignas(128) __bf16 Ks[2][32][64];    // 8 KB
    __shared__ alignas(128) __bf16 Vs[2][64][32];    // 8 KB
    __shared__ alignas(64)  __bf16 Plds[8][16][32];  // 8 KB

    const int tid  = threadIdx.x;
    const int lane = tid & 31;
    const int wave = tid >> 5;
    const int hi = lane >> 4;
    const int ln = lane & 15;

    const int bid = blockIdx.x;
    const int qg = bid & 15;             // 16 q-groups of 128 rows per (b,h)
    const int h  = (bid >> 4) & 15;
    const int b  = bid >> 8;
    const int qBase = qg * 128 + wave * 16;

    const __bf16* qp = Qw + ((size_t)b * HH + h) * TT * KH;
    const __bf16* kp = Kw + ((size_t)b * HH + h) * TT * KH;
    const __bf16* vp = Vt + ((size_t)b * HH + h) * KH * TT;

    // Stage one 32-wide s panel: K 32x64 (256 chunks) + Vt 64x32 (256
    // chunks) = 2 chunks per thread.
    auto issue = [&](int sBase, int buf) {
        {   // K chunk
            const int row = tid >> 3, off = (tid & 7) * 8;
            async_cp16(lds_off(&Ks[buf][row][off]),
                       kp + (size_t)(sBase + row) * KH + off);
        }
        {   // V chunk
            const int row = tid >> 2, off = (tid & 3) * 8;
            async_cp16(lds_off(&Vs[buf][row][off]),
                       vp + (size_t)row * TT + sBase + off);
        }
    };

    issue(0, 0);

    // Q A-fragments (16x32 each), loaded once from global, reused all steps.
    const __bf16* qrow = qp + (size_t)(qBase + ln) * KH;
    v16bf qa0 = load16(qrow + 0  + hi * 16);
    v16bf qa1 = load16(qrow + 32 + hi * 16);

    v8f o[4];
#pragma unroll
    for (int t = 0; t < 4; ++t) o[t] = zero8();
    float mi[8], li[8];
#pragma unroll
    for (int r = 0; r < 8; ++r) { mi[r] = -1e30f; li[r] = 0.f; }

    const int sSteps = (qg + 1) * 4;     // cover s < (qg+1)*128
    for (int ss = 0; ss < sSteps; ++ss) {
        const int sBase = ss << 5;
        const int cur = ss & 1;
        if (ss + 1 < sSteps) { issue((ss + 1) << 5, cur ^ 1); WAIT_ASYNC(2); }
        else                 { WAIT_ASYNC(0); }
        __syncthreads();

        float sv[2][8];
#pragma unroll
        for (int st = 0; st < 2; ++st) {
            const int sB = sBase + st * 16;
            v16bf kb0 = load16(&Ks[cur][st * 16 + ln][0  + hi * 16]);
            v16bf kb1 = load16(&Ks[cur][st * 16 + ln][32 + hi * 16]);
            v8f c = zero8();
            c = wmma_bf16(qa0, kb0, c);
            c = wmma_bf16(qa1, kb1, c);
#pragma unroll
            for (int r = 0; r < 8; ++r) {
                const float x = c[r] * 0.125f;           // 1/sqrt(64)
                const int qi = qBase + r + 8 * hi;
                const int si = sB + ln;
                sv[st][r] = (si > qi) ? -1e30f : x;      // causal mask
            }
        }

        // online softmax; rows r+8*hi live on one 16-lane half-group
        float fac[8];
#pragma unroll
        for (int r = 0; r < 8; ++r) {
            float v = fmaxf(sv[0][r], sv[1][r]);
            v = fmaxf(v, __shfl_xor(v, 1));
            v = fmaxf(v, __shfl_xor(v, 2));
            v = fmaxf(v, __shfl_xor(v, 4));
            v = fmaxf(v, __shfl_xor(v, 8));
            const float nm = fmaxf(mi[r], v);
            fac[r] = __expf(mi[r] - nm);
            mi[r] = nm;
        }
        float rs[8];
#pragma unroll
        for (int r = 0; r < 8; ++r) rs[r] = 0.f;
#pragma unroll
        for (int st = 0; st < 2; ++st) {
#pragma unroll
            for (int r = 0; r < 8; ++r) {
                const float p = __expf(sv[st][r] - mi[r]);
                rs[r] += p;
                Plds[wave][r + 8 * hi][st * 16 + ln] = f2bf(p);
            }
        }
#pragma unroll
        for (int r = 0; r < 8; ++r) {
            float v = rs[r];
            v += __shfl_xor(v, 1);
            v += __shfl_xor(v, 2);
            v += __shfl_xor(v, 4);
            v += __shfl_xor(v, 8);
            li[r] = li[r] * fac[r] + v;
        }
#pragma unroll
        for (int t = 0; t < 4; ++t)
#pragma unroll
            for (int r = 0; r < 8; ++r) o[t][r] *= fac[r];

        // PV: O += P (16x32) @ V (32x64), V panel from LDS
        const __bf16* prow = &Plds[wave][ln][0];
        v16bf pa = load16(prow + hi * 16);
#pragma unroll
        for (int t = 0; t < 4; ++t) {
            v16bf vb = load16(&Vs[cur][t * 16 + ln][hi * 16]);
            o[t] = wmma_bf16(pa, vb, o[t]);
        }
        __syncthreads();
    }

    // normalize + store [B*T, D] bf16 for the output projection
#pragma unroll
    for (int r = 0; r < 8; ++r) li[r] = 1.0f / li[r];
#pragma unroll
    for (int t = 0; t < 4; ++t) {
#pragma unroll
        for (int r = 0; r < 8; ++r) {
            const int trow = qBase + r + 8 * hi;
            const size_t row = (size_t)b * TT + trow;
            const int col = h * 64 + t * 16 + ln;
            Ow[row * DD + col] = f2bf(o[t][r] * li[r]);
        }
    }
}

// ---- kernel 3: output projection  out = O @ Wo^T (fp32 result) ------------

__global__ __launch_bounds__(256)
void out_proj(const __bf16* __restrict__ Ow,
              const __bf16* __restrict__ Wob,
              float* __restrict__ out) {
    __shared__ alignas(128) __bf16 As[2][128][64];
    __shared__ alignas(128) __bf16 Bs[2][64][64];

    const int tid  = threadIdx.x;
    const int lane = tid & 31;
    const int wave = tid >> 5;
    const int hi = lane >> 4;
    const int ln = lane & 15;

    const int mBlock = blockIdx.y * 128;
    const int nBase  = blockIdx.x * 64;

    auto issue = [&](int k0, int buf) {
#pragma unroll
        for (int j = 0; j < 4; ++j) {
            const int c = tid + j * 256;
            const int row = c >> 3, off = (c & 7) * 8;
            async_cp16(lds_off(&As[buf][row][off]),
                       Ow + (size_t)(mBlock + row) * DD + k0 + off);
        }
#pragma unroll
        for (int j = 0; j < 2; ++j) {
            const int c = tid + j * 256;
            const int row = c >> 3, off = (c & 7) * 8;
            async_cp16(lds_off(&Bs[buf][row][off]),
                       Wob + (size_t)(nBase + row) * DD + k0 + off);
        }
    };

    v8f acc[4];
#pragma unroll
    for (int t = 0; t < 4; ++t) acc[t] = zero8();

    issue(0, 0);
    for (int s = 0; s < DD / 64; ++s) {
        const int cur = s & 1;
        if (s + 1 < DD / 64) { issue((s + 1) * 64, cur ^ 1); WAIT_ASYNC(6); }
        else                 { WAIT_ASYNC(0); }
        __syncthreads();
#pragma unroll
        for (int ks = 0; ks < 64; ks += 32) {
            const __bf16* arow = &As[cur][wave * 16 + ln][0];
            v16bf a = load16(arow + ks + hi * 16);
#pragma unroll
            for (int t = 0; t < 4; ++t) {
                v16bf bf = load16(&Bs[cur][t * 16 + ln][ks + hi * 16]);
                acc[t] = wmma_bf16(a, bf, acc[t]);
            }
        }
        __syncthreads();
    }

    const int mBase = mBlock + wave * 16;
#pragma unroll
    for (int t = 0; t < 4; ++t) {
#pragma unroll
        for (int r = 0; r < 8; ++r) {
            const int mG = mBase + r + 8 * hi;
            const int nG = nBase + t * 16 + ln;
            out[(size_t)mG * DD + nG] = acc[t][r];
        }
    }
}

// ---- host launcher --------------------------------------------------------

extern "C" void kernel_launch(void* const* d_in, const int* in_sizes, int n_in,
                              void* d_out, int out_size, void* d_ws, size_t ws_size,
                              hipStream_t stream) {
    const float* x  = (const float*)d_in[0];
    const float* Wq = (const float*)d_in[1];
    const float* Wk = (const float*)d_in[2];
    const float* Wv = (const float*)d_in[3];
    const float* Wo = (const float*)d_in[4];
    float* out = (float*)d_out;

    char* ws = (char*)d_ws;
    size_t off = 0;
    const size_t xElems = (size_t)BB * TT * DD;      // 8Mi
    const size_t wElems = (size_t)DD * DD;           // 1Mi
    __bf16* xb  = (__bf16*)(ws + off); off += xElems * 2;   // 16 MB
    __bf16* wqb = (__bf16*)(ws + off); off += wElems * 2;   //  2 MB
    __bf16* wkb = (__bf16*)(ws + off); off += wElems * 2;
    __bf16* wvb = (__bf16*)(ws + off); off += wElems * 2;
    __bf16* wob = (__bf16*)(ws + off); off += wElems * 2;
    __bf16* Qw  = (__bf16*)(ws + off); off += xElems * 2;   // [B,H,T,64]
    __bf16* Kw  = (__bf16*)(ws + off); off += xElems * 2;   // [B,H,T,64]
    __bf16* Vt  = (__bf16*)(ws + off); off += xElems * 2;   // [B,H,64,T]
    __bf16* Ow  = (__bf16*)(ws + off); off += xElems * 2;   // [B*T, D]
    (void)ws_size; (void)in_sizes; (void)n_in; (void)out_size;

    cast_f32_bf16<<<512, 256, 0, stream>>>(x,  xb,  (int)xElems);
    cast_f32_bf16<<<256, 256, 0, stream>>>(Wq, wqb, (int)wElems);
    cast_f32_bf16<<<256, 256, 0, stream>>>(Wk, wkb, (int)wElems);
    cast_f32_bf16<<<256, 256, 0, stream>>>(Wv, wvb, (int)wElems);
    cast_f32_bf16<<<256, 256, 0, stream>>>(Wo, wob, (int)wElems);

    qkv_proj<<<dim3(16, 64, 3), 256, 0, stream>>>(xb, wqb, wkb, wvb, Qw, Kw, Vt);
    flash_attn<<<1024, 256, 0, stream>>>(Qw, Kw, Vt, Ow);
    out_proj<<<dim3(16, 64, 1), 256, 0, stream>>>(Ow, wob, out);
}
